// InteractConv_22179211116725
// MI455X (gfx1250) — matrix operand order repaired and checked by reading, not compiled, not run on hardware
//
#include <hip/hip_runtime.h>

typedef __attribute__((ext_vector_type(16))) _Float16 v16h;
typedef __attribute__((ext_vector_type(8)))  _Float16 h8;
typedef __attribute__((ext_vector_type(8)))  float    v8f;
typedef __attribute__((ext_vector_type(4)))  float    f4;

#define IN_FEATS 128

static __device__ __forceinline__ v8f wmma_f16(v16h a, v16h b, v8f c) {
  // D = A(16x32 f16) * B(32x16 f16) + C(16x16 f32)
  return __builtin_amdgcn_wmma_f32_16x16x32_f16(
      /*neg_a=*/false, a, /*neg_b=*/false, b,
      /*c_mod=*/(short)0, c, /*reuse_a=*/false, /*reuse_b=*/false);
}

// Pack 16 fp32 (one contiguous run of 16) into a 16-bit A-fragment register.
static __device__ __forceinline__ v16h cvt16(f4 a, f4 b, f4 c, f4 d) {
  v16h v;
  v[0]  = (_Float16)a.x; v[1]  = (_Float16)a.y; v[2]  = (_Float16)a.z; v[3]  = (_Float16)a.w;
  v[4]  = (_Float16)b.x; v[5]  = (_Float16)b.y; v[6]  = (_Float16)b.z; v[7]  = (_Float16)b.w;
  v[8]  = (_Float16)c.x; v[9]  = (_Float16)c.y; v[10] = (_Float16)c.z; v[11] = (_Float16)c.w;
  v[12] = (_Float16)d.x; v[13] = (_Float16)d.y; v[14] = (_Float16)d.z; v[15] = (_Float16)d.w;
  return v;
}

// ---------------------------------------------------------------------------
// Kernel 1: out = bias + feat @ loop_weight   (overwrites d_out)
// ---------------------------------------------------------------------------
__global__ __launch_bounds__(256) void loop_gemm_bias(
    const float* __restrict__ feat, const float* __restrict__ lw,
    const float* __restrict__ bias, float* __restrict__ out, int ntiles) {
  __shared__ __align__(64) _Float16 sLT[128 * 128];  // loop_weight^T [n][k], f16

  int tid = threadIdx.x;
  for (int i = tid; i < 128 * 128; i += 256) {
    int n = i >> 7, k = i & 127;
    sLT[i] = (_Float16)lw[k * 128 + n];
  }
  __syncthreads();

  int lane = tid & 31, wave = tid >> 5;
  int laneM = lane & 15, grp = lane >> 4, off = grp * 8;
  int w = blockIdx.x * 8 + wave, nw = gridDim.x * 8;

  // bias is loop-invariant: preload the 8 values this lane needs
  float bv[8];
#pragma unroll
  for (int nb = 0; nb < 8; ++nb) bv[nb] = bias[nb * 16 + laneM];

  for (int t = w; t < ntiles; t += nw) {
    int n0 = t << 4;
    const float* row = feat + (size_t)(n0 + laneM) * 128;
    v16h A[4];
#pragma unroll
    for (int c = 0; c < 4; ++c) {
      const float* r = row + c * 32;
      A[c] = cvt16(*(const f4*)(r + 2 * off),     *(const f4*)(r + 2 * off + 4),
                   *(const f4*)(r + 2 * off + 8), *(const f4*)(r + 2 * off + 12));
    }
    // one base pointer; all 64 stores use compile-time byte offsets
    float* op = out + (size_t)(n0 + grp * 8) * 128 + laneM;
#pragma unroll
    for (int nbp = 0; nbp < 4; ++nbp) {          // two n-blocks in flight
      v8f acc0 = {0.f,0.f,0.f,0.f,0.f,0.f,0.f,0.f};
      v8f acc1 = {0.f,0.f,0.f,0.f,0.f,0.f,0.f,0.f};
      const _Float16* b0 = sLT + (size_t)(nbp * 32 + laneM) * 128 + grp * 16;
      const _Float16* b1 = b0 + 16 * 128;
#pragma unroll
      for (int c = 0; c < 4; ++c) {
        acc0 = wmma_f16(A[c], *(const v16h*)(b0 + c * 32), acc0);
        acc1 = wmma_f16(A[c], *(const v16h*)(b1 + c * 32), acc1);
      }
#pragma unroll
      for (int r = 0; r < 8; ++r) {
        op[r * 128 + (nbp * 2) * 16]     = acc0[r] + bv[nbp * 2];
        op[r * 128 + (nbp * 2 + 1) * 16] = acc1[r] + bv[nbp * 2 + 1];
      }
    }
  }
}

// ---------------------------------------------------------------------------
// Kernel 2: per-edge 2-layer MLP (WMMA f16) + atomic scatter-add into out
// ---------------------------------------------------------------------------
__global__ __launch_bounds__(256) void edge_mlp_scatter(
    const float* __restrict__ feat, const int* __restrict__ src_idx,
    const int* __restrict__ dst_idx, const float* __restrict__ w1,
    const float* __restrict__ w2, float* __restrict__ out, int ntiles) {
  __shared__ __align__(64) _Float16 sW1T[128 * 256];  // W1^T [n=128][k=256], 64KB
  __shared__ __align__(64) _Float16 sW2T[128 * 128];  // W2^T [n=128][k=128], 32KB
  __shared__ __align__(64) _Float16 sH[8][16 * 128];  // per-wave hidden tile, 32KB
  __shared__ __align__(64) int sDst[8][16];

  int tid = threadIdx.x;
  for (int i = tid; i < 128 * 256; i += 256) {   // W1T[n*256+k] = W1[k*128+n]
    int n = i >> 8, k = i & 255;
    sW1T[i] = (_Float16)w1[k * 128 + n];
  }
  for (int i = tid; i < 128 * 128; i += 256) {   // W2T[n*128+k] = W2[k*128+n]
    int n = i >> 7, k = i & 127;
    sW2T[i] = (_Float16)w2[k * 128 + n];
  }
  __syncthreads();

  int lane = tid & 31, wave = tid >> 5;
  int laneM = lane & 15;   // A-fragment row / B-fragment column
  int grp = lane >> 4;     // lane group (K-half select)
  int off = grp * 8;

  int w = blockIdx.x * 8 + wave, nw = gridDim.x * 8;

  for (int t = w; t < ntiles; t += nw) {
    int e0 = t << 4;
    int sidx = src_idx[e0 + laneM];
    int didx = dst_idx[e0 + laneM];
    if (grp == 0) sDst[wave][laneM] = didx;
    const float* srow = feat + (size_t)sidx * 128;
    const float* drow = feat + (size_t)didx * 128;

    // Gather concat(feat[src], feat[dst]) as 8 A-fragments (16x256 f16 tile)
    v16h A[8];
#pragma unroll
    for (int c = 0; c < 8; ++c) {
      const float* r = (c < 4) ? (srow + c * 32) : (drow + (c - 4) * 32);
      A[c] = cvt16(*(const f4*)(r + 2 * off),     *(const f4*)(r + 2 * off + 4),
                   *(const f4*)(r + 2 * off + 8), *(const f4*)(r + 2 * off + 12));
    }

    // Layer 1: H = relu(A @ W1); two independent WMMA chains per pass so the
    // scheduler fills f16-WMMA RAW hazard slots with real work.
#pragma unroll
    for (int nbp = 0; nbp < 4; ++nbp) {
      v8f acc0 = {0.f,0.f,0.f,0.f,0.f,0.f,0.f,0.f};
      v8f acc1 = {0.f,0.f,0.f,0.f,0.f,0.f,0.f,0.f};
      const _Float16* b0 = sW1T + (size_t)(nbp * 32 + laneM) * 256 + grp * 16;
      const _Float16* b1 = b0 + 16 * 256;
#pragma unroll
      for (int c = 0; c < 8; ++c) {
        acc0 = wmma_f16(A[c], *(const v16h*)(b0 + c * 32), acc0);
        acc1 = wmma_f16(A[c], *(const v16h*)(b1 + c * 32), acc1);
      }
      _Float16* hp0 = &sH[wave][0] + (nbp * 2) * 16 + laneM;
#pragma unroll
      for (int r = 0; r < 8; ++r) {
        float v0 = acc0[r]; v0 = v0 > 0.f ? v0 : 0.f;
        float v1 = acc1[r]; v1 = v1 > 0.f ? v1 : 0.f;
        hp0[(grp * 8 + r) * 128]      = (_Float16)v0;  // ds stores; in-order per wave
        hp0[(grp * 8 + r) * 128 + 16] = (_Float16)v1;
      }
    }

    // Re-load H as A-fragments (per-wave LDS, DS ops in-order: no barrier).
    v16h H[4];
#pragma unroll
    for (int c = 0; c < 4; ++c) {
      const _Float16* r = &sH[wave][laneM * 128 + c * 32];
      h8 lo = *(const h8*)(r + 2 * off);
      h8 hi = *(const h8*)(r + 2 * off + 8);
      H[c] = __builtin_shufflevector(lo, hi, 0, 1, 2, 3, 4, 5, 6, 7,
                                     8, 9, 10, 11, 12, 13, 14, 15);
    }

    // Scatter indices: two int4 LDS vector loads, then 8 base pointers once.
    int4 d0 = *(const int4*)&sDst[wave][grp * 8];
    int4 d1 = *(const int4*)&sDst[wave][grp * 8 + 4];
    float* bp[8];
    bp[0] = out + (size_t)d0.x * 128 + laneM;
    bp[1] = out + (size_t)d0.y * 128 + laneM;
    bp[2] = out + (size_t)d0.z * 128 + laneM;
    bp[3] = out + (size_t)d0.w * 128 + laneM;
    bp[4] = out + (size_t)d1.x * 128 + laneM;
    bp[5] = out + (size_t)d1.y * 128 + laneM;
    bp[6] = out + (size_t)d1.z * 128 + laneM;
    bp[7] = out + (size_t)d1.w * 128 + laneM;

    // Layer 2 + scatter-add (L2-resident fp32 atomics, immediate col offsets)
#pragma unroll
    for (int nbp = 0; nbp < 4; ++nbp) {
      v8f acc0 = {0.f,0.f,0.f,0.f,0.f,0.f,0.f,0.f};
      v8f acc1 = {0.f,0.f,0.f,0.f,0.f,0.f,0.f,0.f};
      const _Float16* b0 = sW2T + (size_t)(nbp * 32 + laneM) * 128 + grp * 16;
      const _Float16* b1 = b0 + 16 * 128;
#pragma unroll
      for (int c = 0; c < 4; ++c) {
        acc0 = wmma_f16(H[c], *(const v16h*)(b0 + c * 32), acc0);
        acc1 = wmma_f16(H[c], *(const v16h*)(b1 + c * 32), acc1);
      }
#pragma unroll
      for (int r = 0; r < 8; ++r) {
        unsafeAtomicAdd(bp[r] + (nbp * 2) * 16,     acc0[r]);
        unsafeAtomicAdd(bp[r] + (nbp * 2 + 1) * 16, acc1[r]);
      }
    }
  }
}

extern "C" void kernel_launch(void* const* d_in, const int* in_sizes, int n_in,
                              void* d_out, int out_size, void* d_ws, size_t ws_size,
                              hipStream_t stream) {
  const float* feat    = (const float*)d_in[0];
  const int*   src_idx = (const int*)d_in[1];
  const int*   dst_idx = (const int*)d_in[2];
  const float* w1      = (const float*)d_in[3];
  const float* w2      = (const float*)d_in[4];
  const float* lw      = (const float*)d_in[5];
  const float* bias    = (const float*)d_in[6];
  float* out = (float*)d_out;

  int nNodes = in_sizes[0] / IN_FEATS;   // 50000
  int nEdges = in_sizes[1];              // 800000
  int node_tiles = nNodes / 16;          // 3125 (exact)
  int edge_tiles = nEdges / 16;          // 50000 (exact)

  // 1) out = bias + feat @ loop_weight  (initializes d_out)
  hipLaunchKernelGGL(loop_gemm_bias, dim3(391), dim3(256), 0, stream,
                     feat, lw, bias, out, node_tiles);
  // 2) edge MLP + atomic scatter accumulate (same stream -> ordered)
  hipLaunchKernelGGL(edge_mlp_scatter, dim3(625), dim3(256), 0, stream,
                     feat, src_idx, dst_idx, w1, w2, out, edge_tiles);
}
